// ParallelSelfAttention_8967891714621
// MI455X (gfx1250) — matrix operand order repaired and checked by reading, not compiled, not run on hardware
//
#include <hip/hip_runtime.h>
#include <hip/hip_bf16.h>

#define S_LEN 1024
#define BATCH 4
#define HID 2048
#define NHEADS 16
#define HDIM 128
#define N_QKV (3 * HID)

typedef __attribute__((ext_vector_type(16))) __bf16 v16bf;
typedef __attribute__((ext_vector_type(8)))  float  v8f;
typedef __attribute__((ext_vector_type(8)))  unsigned int v8u;
typedef __attribute__((ext_vector_type(4)))  unsigned int u32x4;
typedef __attribute__((ext_vector_type(8)))  int i32x8;
typedef __attribute__((ext_vector_type(4)))  int i32x4;

#if defined(__has_builtin)
#if __has_builtin(__builtin_amdgcn_tensor_load_to_lds) && \
    __has_builtin(__builtin_amdgcn_s_wait_tensorcnt)
#define USE_TDM 1
#endif
#endif
#ifndef USE_TDM
#define USE_TDM 0
#endif

__device__ __forceinline__ unsigned int pack2bf(float lo, float hi) {
  unsigned int a = __float_as_uint(lo);
  unsigned int b = __float_as_uint(hi);
  a += 0x7FFFu + ((a >> 16) & 1u);   // round-to-nearest-even
  b += 0x7FFFu + ((b >> 16) & 1u);
  return (a >> 16) | (b & 0xFFFF0000u);
}
__device__ __forceinline__ unsigned short f2bf(float f) {
  unsigned int a = __float_as_uint(f);
  a += 0x7FFFu + ((a >> 16) & 1u);
  return (unsigned short)(a >> 16);
}
__device__ __forceinline__ v8f wmma_bf16(v8u a, v8u b, v8f c) {
  union { v8u u; v16bf h; } A, B;
  A.u = a; B.u = b;
  return __builtin_amdgcn_wmma_f32_16x16x32_bf16(
      false, A.h, false, B.h, (short)0, c, false, false);
}
__device__ __forceinline__ float redmax16(float v) {
  v = fmaxf(v, __shfl_xor(v, 1));
  v = fmaxf(v, __shfl_xor(v, 2));
  v = fmaxf(v, __shfl_xor(v, 4));
  v = fmaxf(v, __shfl_xor(v, 8));
  return v;
}
__device__ __forceinline__ float redsum16(float v) {
  v += __shfl_xor(v, 1);
  v += __shfl_xor(v, 2);
  v += __shfl_xor(v, 4);
  v += __shfl_xor(v, 8);
  return v;
}

// ---------------------------------------------------------------------------
// Kernel 1: fused QKV projection. A = hidden_states [S*B, H] fp32,
// B = W_qkv [H, 3H] fp32. Output scattered as bf16 into Q/K/V [B,NH,S,HD].
// Q is pre-scaled by 1/sqrt(HD).
// ---------------------------------------------------------------------------
__global__ __launch_bounds__(256) void qkv_wmma_kernel(
    const float* __restrict__ hs, const float* __restrict__ Wqkv,
    const float* __restrict__ bqkv,
    unsigned short* __restrict__ Qb, unsigned short* __restrict__ Kb,
    unsigned short* __restrict__ Vb) {
  __shared__ unsigned int Alds[64][17];   // [row][k-pair], padded
  __shared__ unsigned int Blds[16][132];  // [k-pair][n], padded

  const int tid = threadIdx.x;
  const int wave = tid >> 5, lane = tid & 31;
  const int m = lane & 15, half = lane >> 4;
  const int rg = wave & 3, cg0 = (wave >> 2) << 2;
  const int mblock = blockIdx.y * 64;
  const int nblock = blockIdx.x * 128;

  v8f acc[4] = {};

  for (int k0 = 0; k0 < HID; k0 += 32) {
    if (k0 + 32 < HID) {  // prefetch next K-step tiles into cache
      __builtin_prefetch(hs + (size_t)(mblock + (tid >> 4)) * HID + k0 + 32, 0, 1);
      __builtin_prefetch(Wqkv + (size_t)(k0 + 32 + (tid >> 7)) * N_QKV + nblock + (tid & 127), 0, 1);
    }
    __syncthreads();
#pragma unroll
    for (int i = 0; i < 4; ++i) {  // A tile 64x32 fp32 -> 64x16 bf16-pairs
      int lin = tid + i * 256;
      int row = lin >> 4, p = lin & 15;
      const float* src = hs + (size_t)(mblock + row) * HID + k0 + p * 2;
      Alds[row][p] = pack2bf(src[0], src[1]);
    }
#pragma unroll
    for (int i = 0; i < 8; ++i) {  // B tile 32x128 fp32 -> 16x128 bf16-pairs
      int lin = tid + i * 256;
      int p = lin >> 7, n = lin & 127;
      const float* src = Wqkv + (size_t)(k0 + 2 * p) * N_QKV + nblock + n;
      Blds[p][n] = pack2bf(src[0], src[N_QKV]);
    }
    __syncthreads();

    v8u a;
#pragma unroll
    for (int j = 0; j < 8; ++j)
      a[j] = Alds[rg * 16 + m][(j >> 2) * 8 + half * 4 + (j & 3)];
#pragma unroll
    for (int t = 0; t < 4; ++t) {
      v8u bb;
#pragma unroll
      for (int j = 0; j < 8; ++j)
        bb[j] = Blds[(j >> 2) * 8 + half * 4 + (j & 3)][(cg0 + t) * 16 + m];
      acc[t] = wmma_bf16(a, bb, acc[t]);
    }
  }

  const float invnorm = 0.08838834764831845f;  // 1/sqrt(128)
#pragma unroll
  for (int t = 0; t < 4; ++t) {
    int n = nblock + (cg0 + t) * 16 + m;
    int head = n / (3 * HDIM);
    int rem = n % (3 * HDIM);
    int which = rem / HDIM;
    int d = rem % HDIM;
    float bias = bqkv[n];
    unsigned short* dst = (which == 0) ? Qb : (which == 1) ? Kb : Vb;
    float scale = (which == 0) ? invnorm : 1.0f;
#pragma unroll
    for (int r = 0; r < 8; ++r) {
      int mg = mblock + rg * 16 + r + 8 * half;  // row = s*B + b
      int s = mg / BATCH, b = mg % BATCH;
      float v = (acc[t][r] + bias) * scale;
      dst[((size_t)((b * NHEADS + head) * S_LEN + s)) * HDIM + d] = f2bf(v);
    }
  }
}

// ---------------------------------------------------------------------------
// Kernel 2: causal flash attention with ALiBi. One block = (b,h) x 128 queries.
// 8 waves, each owns a 16-query strip; online softmax in fp32.
// K tiles staged by the Tensor Data Mover (TDM) when available; its LDS
// padding feature (64 DWORDs -> +4 DWORDs) reproduces the [32][68] layout.
// ---------------------------------------------------------------------------
__global__ __launch_bounds__(256) void attn_wmma_kernel(
    const unsigned short* __restrict__ Qb, const unsigned short* __restrict__ Kb,
    const unsigned short* __restrict__ Vb, unsigned short* __restrict__ Cb) {
  __shared__ unsigned int Klds[32][68];      // [key][d-pair], padded
  __shared__ unsigned int Vtlds[128][20];    // [d][k-pair], transposed, padded
  __shared__ unsigned short Plds[8][16][32]; // per-wave P tile (C->A layout hop)

  const int tid = threadIdx.x;
  const int wave = tid >> 5, lane = tid & 31;
  const int m = lane & 15, half = lane >> 4;
  const int bh = blockIdx.y;
  const int b = bh / NHEADS, h = bh % NHEADS;
  const int qbase = blockIdx.x * 128;
  const int qb0 = qbase + wave * 16;
  const float slope = exp2f(-0.5f * (float)(h + 1));  // 2^(-8(h+1)/16)

  const unsigned int* Qu = (const unsigned int*)Qb + (size_t)(bh * S_LEN) * 64;
  const unsigned int* Ku = (const unsigned int*)Kb + (size_t)(bh * S_LEN) * 64;
  const unsigned short* Vp = Vb + (size_t)(bh * S_LEN) * HDIM;

#if USE_TDM
  const unsigned lds_k_off = (unsigned)(uintptr_t)(&Klds[0][0]);
#endif

  // Q fragments for this wave's 16 rows (4 d-chunks of 32)
  v8u qf[4];
  const unsigned int* qrow = Qu + (size_t)(qb0 + m) * 64;
#pragma unroll
  for (int c = 0; c < 4; ++c)
#pragma unroll
    for (int j = 0; j < 8; ++j)
      qf[c][j] = qrow[c * 16 + (j >> 2) * 8 + half * 4 + (j & 3)];

  float rmax[8], rsum[8];
  v8f O[8] = {};
#pragma unroll
  for (int r = 0; r < 8; ++r) { rmax[r] = -1e30f; rsum[r] = 0.0f; }

  const int nkt = (qbase + 128) >> 5;  // causal: keys <= qbase+127
  for (int kt = 0; kt < nkt; ++kt) {
    const int kbase = kt << 5;
    if (kt + 1 < nkt) {  // prefetch next tile
      __builtin_prefetch(Ku + (size_t)(kbase + 32 + (tid >> 6)) * 64 + (tid & 63), 0, 1);
      __builtin_prefetch(Vp + (size_t)(kbase + 32 + (tid >> 7)) * HDIM + (tid & 127), 0, 1);
    }
    __syncthreads();
#if USE_TDM
    if (wave == 0) {  // one descriptor moves the whole 8KB K tile, padded
      unsigned long long ga =
          (unsigned long long)(uintptr_t)(Ku + (size_t)kbase * 64);
      u32x4 g0;
      g0[0] = 1u;                                   // count=1, user mode
      g0[1] = lds_k_off;                            // lds_addr (bytes)
      g0[2] = (unsigned)(ga & 0xFFFFFFFFu);         // global_addr[31:0]
      g0[3] = (unsigned)((ga >> 32) & 0x01FFFFFFu)  // global_addr[56:32]
              | 0x80000000u;                        // type = 2 ("image")
      i32x8 g1;
      g1[0] = (int)(0x30000u        // data_size = 8B
                    | 0x100000u     // pad_enable
                    | (5u << 22)    // pad_interval: 64 DWORDs
                    | (3u << 25));  // pad_amount: 4 DWORDs
      g1[1] = (int)(32u << 16);     // tensor_dim0 = 32 units (256B row)
      g1[2] = (int)(0xFFFFu << 16); // tensor_dim1 (OOB bound, generous)
      g1[3] = (int)(32u << 16);     // tile_dim0 = 32 units
      g1[4] = 32;                   // tile_dim1 = 32 rows
      g1[5] = 32;                   // tensor_dim0_stride = 32 units
      g1[6] = 0;
      g1[7] = 0;
      i32x4 gz = {};
#if defined(__clang_major__) && __clang_major__ >= 23
      i32x8 gz8 = {};
      __builtin_amdgcn_tensor_load_to_lds(g0, g1, gz, gz, gz8, 0);
#else
      __builtin_amdgcn_tensor_load_to_lds(g0, g1, gz, gz, 0);
#endif
    }
#else
#pragma unroll
    for (int i = 0; i < 8; ++i) {  // K tile 32 keys x 128 d (as 64 uints)
      int lin = tid + i * 256;
      int key = lin >> 6, dp = lin & 63;
      Klds[key][dp] = Ku[(size_t)(kbase + key) * 64 + dp];
    }
#endif
#pragma unroll
    for (int i = 0; i < 8; ++i) {  // V tile transposed: Vt[d][k-pair]
      int lin = tid + i * 256;
      int d = lin & 127, kp = lin >> 7;
      unsigned int lo = Vp[(size_t)(kbase + 2 * kp) * HDIM + d];
      unsigned int hi = Vp[(size_t)(kbase + 2 * kp + 1) * HDIM + d];
      Vtlds[d][kp] = lo | (hi << 16);
    }
#if USE_TDM
    if (wave == 0) __builtin_amdgcn_s_wait_tensorcnt(0);
#endif
    __syncthreads();
    if (kbase > qb0 + 15) continue;  // wave-uniform: beyond diagonal

    // scores: 16 queries x 32 keys = two C tiles, K accumulated over d=128
    v8f st[2] = {};
#pragma unroll
    for (int g = 0; g < 2; ++g)
#pragma unroll
      for (int c = 0; c < 4; ++c) {
        v8u bb;
#pragma unroll
        for (int j = 0; j < 8; ++j)
          bb[j] = Klds[g * 16 + m][c * 16 + (j >> 2) * 8 + half * 4 + (j & 3)];
        st[g] = wmma_bf16(qf[c], bb, st[g]);
      }

    // online softmax (rows live across 16-lane halves; reduce with shfl_xor)
#pragma unroll
    for (int r = 0; r < 8; ++r) {
      int irow = qb0 + r + 8 * half;
      int j0 = kbase + m, j1 = kbase + 16 + m;
      float s0 = st[0][r] + slope * (float)(j0 - irow);
      float s1 = st[1][r] + slope * (float)(j1 - irow);
      if (j0 > irow) s0 = -1e9f;
      if (j1 > irow) s1 = -1e9f;
      float mx = redmax16(fmaxf(s0, s1));
      float mnew = fmaxf(rmax[r], mx);
      float corr = __expf(rmax[r] - mnew);
      rmax[r] = mnew;
      float p0 = __expf(s0 - mnew);
      float p1 = __expf(s1 - mnew);
      rsum[r] = rsum[r] * corr + redsum16(p0 + p1);
#pragma unroll
      for (int d = 0; d < 8; ++d) O[d][r] *= corr;
      Plds[wave][r + 8 * half][m] = f2bf(p0);
      Plds[wave][r + 8 * half][16 + m] = f2bf(p1);
    }

    // P (16x32) back as A fragment; PV accumulates 16x128 output
    const unsigned int* Pu = (const unsigned int*)&Plds[wave][m][0];
    v8u pf;
#pragma unroll
    for (int j = 0; j < 8; ++j)
      pf[j] = Pu[(j >> 2) * 8 + half * 4 + (j & 3)];
#pragma unroll
    for (int d = 0; d < 8; ++d) {
      v8u bb;
#pragma unroll
      for (int j = 0; j < 8; ++j)
        bb[j] = Vtlds[d * 16 + m][(j >> 2) * 8 + half * 4 + (j & 3)];
      O[d] = wmma_bf16(pf, bb, O[d]);
    }
  }

  // finalize: O / rowsum -> context [S,B,H] bf16
#pragma unroll
  for (int r = 0; r < 8; ++r) rsum[r] = 1.0f / rsum[r];
#pragma unroll
  for (int d = 0; d < 8; ++d) {
    int col = h * HDIM + d * 16 + m;
#pragma unroll
    for (int r = 0; r < 8; ++r) {
      int srow = qb0 + r + 8 * half;
      Cb[(size_t)(srow * BATCH + b) * HID + col] = f2bf(O[d][r] * rsum[r]);
    }
  }
}

// ---------------------------------------------------------------------------
// Kernel 3: dense projection. A = context bf16 [S*B, H], B = W_dense fp32.
// ---------------------------------------------------------------------------
__global__ __launch_bounds__(256) void dense_wmma_kernel(
    const unsigned short* __restrict__ Cb, const float* __restrict__ Wd,
    float* __restrict__ out) {
  __shared__ unsigned int Alds[64][17];
  __shared__ unsigned int Blds[16][132];
  const int tid = threadIdx.x;
  const int wave = tid >> 5, lane = tid & 31;
  const int m = lane & 15, half = lane >> 4;
  const int rg = wave & 3, cg0 = (wave >> 2) << 2;
  const int mblock = blockIdx.y * 64;
  const int nblock = blockIdx.x * 128;
  const unsigned int* Cu = (const unsigned int*)Cb;

  v8f acc[4] = {};
  for (int k0 = 0; k0 < HID; k0 += 32) {
    if (k0 + 32 < HID) {
      __builtin_prefetch(Cu + (size_t)(mblock + (tid >> 4)) * (HID / 2) + ((k0 + 32) >> 1), 0, 1);
      __builtin_prefetch(Wd + (size_t)(k0 + 32 + (tid >> 7)) * HID + nblock + (tid & 127), 0, 1);
    }
    __syncthreads();
#pragma unroll
    for (int i = 0; i < 4; ++i) {
      int lin = tid + i * 256;
      int row = lin >> 4, p = lin & 15;
      Alds[row][p] = Cu[(size_t)(mblock + row) * (HID / 2) + (k0 >> 1) + p];
    }
#pragma unroll
    for (int i = 0; i < 8; ++i) {
      int lin = tid + i * 256;
      int p = lin >> 7, n = lin & 127;
      const float* src = Wd + (size_t)(k0 + 2 * p) * HID + nblock + n;
      Blds[p][n] = pack2bf(src[0], src[HID]);
    }
    __syncthreads();
    v8u a;
#pragma unroll
    for (int j = 0; j < 8; ++j)
      a[j] = Alds[rg * 16 + m][(j >> 2) * 8 + half * 4 + (j & 3)];
#pragma unroll
    for (int t = 0; t < 4; ++t) {
      v8u bb;
#pragma unroll
      for (int j = 0; j < 8; ++j)
        bb[j] = Blds[(j >> 2) * 8 + half * 4 + (j & 3)][(cg0 + t) * 16 + m];
      acc[t] = wmma_bf16(a, bb, acc[t]);
    }
  }
#pragma unroll
  for (int t = 0; t < 4; ++t) {
    int n = nblock + (cg0 + t) * 16 + m;
#pragma unroll
    for (int r = 0; r < 8; ++r) {
      int mg = mblock + rg * 16 + r + 8 * half;
      out[(size_t)mg * HID + n] = acc[t][r];
    }
  }
}

__global__ void bias_copy_kernel(const float* __restrict__ bd,
                                 float* __restrict__ out) {
  int i = blockIdx.x * blockDim.x + threadIdx.x;
  if (i < HID) out[i] = bd[i];
}

extern "C" void kernel_launch(void* const* d_in, const int* in_sizes, int n_in,
                              void* d_out, int out_size, void* d_ws, size_t ws_size,
                              hipStream_t stream) {
  const float* hs   = (const float*)d_in[0];
  const float* Wqkv = (const float*)d_in[1];
  const float* bqkv = (const float*)d_in[2];
  const float* Wd   = (const float*)d_in[3];
  const float* bd   = (const float*)d_in[4];
  // d_in[5] = attention_mask (causal, implemented analytically)
  // d_in[6] = position_ids (unused by reference math)

  const size_t per = (size_t)BATCH * NHEADS * S_LEN * HDIM;  // 8M elems
  unsigned short* Qb = (unsigned short*)d_ws;
  unsigned short* Kb = Qb + per;
  unsigned short* Vb = Kb + per;
  unsigned short* Cb = Vb + per;
  float* out = (float*)d_out;

  qkv_wmma_kernel<<<dim3(N_QKV / 128, (S_LEN * BATCH) / 64), 256, 0, stream>>>(
      hs, Wqkv, bqkv, Qb, Kb, Vb);
  attn_wmma_kernel<<<dim3(S_LEN / 128, BATCH * NHEADS), 256, 0, stream>>>(
      Qb, Kb, Vb, Cb);
  dense_wmma_kernel<<<dim3(HID / 128, (S_LEN * BATCH) / 64), 256, 0, stream>>>(
      Cb, Wd, out);
  bias_copy_kernel<<<dim3(HID / 256), 256, 0, stream>>>(
      bd, out + (size_t)S_LEN * BATCH * HID);
}
